// Message_79345225826318
// MI455X (gfx1250) — matrix operand order repaired and checked by reading, not compiled, not run on hardware
//
#include <hip/hip_runtime.h>
#include <hip/hip_bf16.h>

#define NN  50000
#define EE  800000
#define FF  64
#define FEE 32
#define HH  128
#define DIN 160
#define EPB 64      // edges per block
#define SA  168     // LDS stride for A (160 + 8 pad), keeps 16B alignment
#define SH  136     // LDS stride for H1/H2 (128 + 8 pad)

typedef __attribute__((ext_vector_type(16))) __bf16 v16bf;
typedef __attribute__((ext_vector_type(8)))  float  v8f;

union Frag {
    v16bf v;
    uint4 u[2];
};

__device__ __forceinline__ unsigned short f2bf(float f) {
    unsigned u = __float_as_uint(f);
    u += 0x7FFFu + ((u >> 16) & 1u);   // round-to-nearest-even
    return (unsigned short)(u >> 16);
}

__device__ __forceinline__ int clampN(int v) {
    return v < 0 ? 0 : (v >= NN ? NN - 1 : v);
}

// ---------------------------------------------------------------------------
// Prep: zero accumulators, swizzle W1/W2/W3 (fp32 row-major) into per-lane
// WMMA B-fragment order: flat = (((kk*strips + ns)*32 + lane)*16 + e),
// element = W[K= kk*32 + (lane/16)*16 + e][N = ns*16 + lane%16] as bf16.
// ---------------------------------------------------------------------------
__global__ __launch_bounds__(256) void prep_kernel(
    const float* __restrict__ W1, const float* __restrict__ W2,
    const float* __restrict__ W3,
    unsigned short* __restrict__ W1s, unsigned short* __restrict__ W2s,
    unsigned short* __restrict__ W3s,
    float* __restrict__ sums, float* __restrict__ counts)
{
    int i = blockIdx.x * 256 + threadIdx.x;
    if (i < NN * FF) sums[i] = 0.0f;
    if (i < NN)      counts[i] = 0.0f;
    if (i < DIN * HH) {
        int e = i & 15, lane = (i >> 4) & 31, ns = (i >> 9) & 7, kk = i >> 12;
        int K  = kk * 32 + ((lane >> 4) << 4) + e;
        int Nc = ns * 16 + (lane & 15);
        W1s[i] = f2bf(W1[K * HH + Nc]);
    }
    if (i < HH * HH) {
        int e = i & 15, lane = (i >> 4) & 31, ns = (i >> 9) & 7, kk = i >> 12;
        int K  = kk * 32 + ((lane >> 4) << 4) + e;
        int Nc = ns * 16 + (lane & 15);
        W2s[i] = f2bf(W2[K * HH + Nc]);
    }
    if (i < HH * FF) {
        int e = i & 15, lane = (i >> 4) & 31, ns = (i >> 9) & 3, kk = i >> 11;
        int K  = kk * 32 + ((lane >> 4) << 4) + e;
        int Nc = ns * 16 + (lane & 15);
        W3s[i] = f2bf(W3[K * FF + Nc]);
    }
}

// ---------------------------------------------------------------------------
// Edge MLP: 64 edges/block, 8 waves. bf16 WMMA for all three GEMMs,
// fp32 atomic scatter of messages.
// ---------------------------------------------------------------------------
__global__ __launch_bounds__(256, 1) void edge_mlp_kernel(
    const float* __restrict__ x,  const int* __restrict__ ei,
    const float* __restrict__ ef,
    const unsigned short* __restrict__ W1s, const float* __restrict__ b1,
    const unsigned short* __restrict__ W2s, const float* __restrict__ b2,
    const unsigned short* __restrict__ W3s, const float* __restrict__ b3,
    float* __restrict__ sums, float* __restrict__ counts)
{
    __shared__ unsigned short ldsA[EPB * SA];   // A (64x160); later aliased as H2 (stride SH)
    __shared__ unsigned short ldsH1[EPB * SH];  // H1 (64x128)

    const int tid   = threadIdx.x;
    const int wave  = tid >> 5;
    const int lane  = tid & 31;
    const int lrow  = lane & 15;
    const int lhalf = lane >> 4;
    const int e0    = blockIdx.x * EPB;

    // ---- gather + concat + fp32->bf16, build A tile in LDS ----
    for (int i = tid; i < EPB * 40; i += 256) {
        int e  = i / 40;
        int c4 = i - e * 40;
        const float* p;
        if (c4 < 16) {
            int s = clampN(ei[e0 + e]);
            p = x + s * FF + c4 * 4;
        } else if (c4 < 32) {
            int t = clampN(ei[EE + e0 + e]);
            p = x + t * FF + (c4 - 16) * 4;
        } else {
            p = ef + (e0 + e) * FEE + (c4 - 32) * 4;
        }
        float4 v = *(const float4*)p;
        uint2 pk;
        pk.x = (unsigned)f2bf(v.x) | ((unsigned)f2bf(v.y) << 16);
        pk.y = (unsigned)f2bf(v.z) | ((unsigned)f2bf(v.w) << 16);
        *(uint2*)&ldsA[e * SA + c4 * 4] = pk;
    }
    if (tid < EPB) {
        int t = clampN(ei[EE + e0 + tid]);
        atomicAdd(&counts[t], 1.0f);
    }
    __syncthreads();

    // ---- GEMM1: H1 = relu(A @ W1 + b1), wave owns N-strip `wave` ----
    {
        v8f acc[4] = {};
        #pragma unroll
        for (int kk = 0; kk < 5; ++kk) {
            Frag bfr;
            const uint4* wp = (const uint4*)(W1s + (((kk << 3) + wave) * 32 + lane) * 16);
            bfr.u[0] = wp[0];
            bfr.u[1] = wp[1];
            const int kb = kk * 32 + (lhalf << 4);
            #pragma unroll
            for (int rt = 0; rt < 4; ++rt) {
                Frag af;
                const unsigned short* ap = &ldsA[(rt * 16 + lrow) * SA + kb];
                af.u[0] = *(const uint4*)ap;
                af.u[1] = *(const uint4*)(ap + 8);
                acc[rt] = __builtin_amdgcn_wmma_f32_16x16x32_bf16(
                    false, af.v, false, bfr.v, (short)0, acc[rt], false, false);
            }
        }
        const int col = (wave << 4) + lrow;
        const float bias = b1[col];
        #pragma unroll
        for (int rt = 0; rt < 4; ++rt)
            #pragma unroll
            for (int r = 0; r < 8; ++r) {
                float vv = fmaxf(acc[rt][r] + bias, 0.0f);
                ldsH1[(rt * 16 + (lhalf << 3) + r) * SH + col] = f2bf(vv);
            }
    }
    __syncthreads();

    // ---- GEMM2: H2 = relu(H1 @ W2 + b2), write H2 over A buffer ----
    {
        v8f acc[4] = {};
        #pragma unroll
        for (int kk = 0; kk < 4; ++kk) {
            Frag bfr;
            const uint4* wp = (const uint4*)(W2s + (((kk << 3) + wave) * 32 + lane) * 16);
            bfr.u[0] = wp[0];
            bfr.u[1] = wp[1];
            const int kb = kk * 32 + (lhalf << 4);
            #pragma unroll
            for (int rt = 0; rt < 4; ++rt) {
                Frag af;
                const unsigned short* ap = &ldsH1[(rt * 16 + lrow) * SH + kb];
                af.u[0] = *(const uint4*)ap;
                af.u[1] = *(const uint4*)(ap + 8);
                acc[rt] = __builtin_amdgcn_wmma_f32_16x16x32_bf16(
                    false, af.v, false, bfr.v, (short)0, acc[rt], false, false);
            }
        }
        const int col = (wave << 4) + lrow;
        const float bias = b2[col];
        #pragma unroll
        for (int rt = 0; rt < 4; ++rt)
            #pragma unroll
            for (int r = 0; r < 8; ++r) {
                float vv = fmaxf(acc[rt][r] + bias, 0.0f);
                ldsA[(rt * 16 + (lhalf << 3) + r) * SH + col] = f2bf(vv);
            }
    }
    __syncthreads();

    // ---- GEMM3: MSG = H2 @ W3 + b3, fp32 atomic scatter to sums[target] ----
    {
        const int rt    = wave & 3;   // row tile (16 edges)
        const int chalf = wave >> 2;  // column-tile pair
        v8f acc[2] = {};
        #pragma unroll
        for (int kk = 0; kk < 4; ++kk) {
            Frag af;
            const unsigned short* ap = &ldsA[(rt * 16 + lrow) * SH + kk * 32 + (lhalf << 4)];
            af.u[0] = *(const uint4*)ap;
            af.u[1] = *(const uint4*)(ap + 8);
            #pragma unroll
            for (int c = 0; c < 2; ++c) {
                const int ct = (chalf << 1) + c;
                Frag bfr;
                const uint4* wp = (const uint4*)(W3s + (((kk << 2) + ct) * 32 + lane) * 16);
                bfr.u[0] = wp[0];
                bfr.u[1] = wp[1];
                acc[c] = __builtin_amdgcn_wmma_f32_16x16x32_bf16(
                    false, af.v, false, bfr.v, (short)0, acc[c], false, false);
            }
        }
        int tg[8];
        #pragma unroll
        for (int r = 0; r < 8; ++r)
            tg[r] = clampN(ei[EE + e0 + rt * 16 + (lhalf << 3) + r]);
        #pragma unroll
        for (int c = 0; c < 2; ++c) {
            const int col = ((chalf << 1) + c) * 16 + lrow;
            const float bias = b3[col];
            #pragma unroll
            for (int r = 0; r < 8; ++r)
                atomicAdd(&sums[tg[r] * FF + col], acc[c][r] + bias);
        }
    }
}

// ---------------------------------------------------------------------------
// Finalize: out = x + sums / max(counts, 1)
// ---------------------------------------------------------------------------
__global__ __launch_bounds__(256) void finalize_kernel(
    const float* __restrict__ x, const float* __restrict__ sums,
    const float* __restrict__ counts, float* __restrict__ out)
{
    int i = blockIdx.x * 256 + threadIdx.x;
    if (i >= NN * FF) return;
    float c = fmaxf(counts[i >> 6], 1.0f);
    out[i] = x[i] + sums[i] / c;
}

extern "C" void kernel_launch(void* const* d_in, const int* in_sizes, int n_in,
                              void* d_out, int out_size, void* d_ws, size_t ws_size,
                              hipStream_t stream) {
    const float* x  = (const float*)d_in[0];
    const int*   ei = (const int*)  d_in[1];
    const float* ef = (const float*)d_in[2];
    const float* W1 = (const float*)d_in[3];
    const float* b1 = (const float*)d_in[4];
    const float* W2 = (const float*)d_in[5];
    const float* b2 = (const float*)d_in[6];
    const float* W3 = (const float*)d_in[7];
    const float* b3 = (const float*)d_in[8];

    char* ws = (char*)d_ws;
    unsigned short* W1s = (unsigned short*)(ws);                    // 40960 B
    unsigned short* W2s = (unsigned short*)(ws + 40960);            // 32768 B
    unsigned short* W3s = (unsigned short*)(ws + 40960 + 32768);    // 16384 B
    float* sums   = (float*)(ws + 90112);                           // N*F*4 B
    float* counts = (float*)(ws + 90112 + (size_t)NN * FF * 4);     // N*4 B

    const int pblocks = (NN * FF + 255) / 256;  // 12500
    prep_kernel<<<pblocks, 256, 0, stream>>>(W1, W2, W3, W1s, W2s, W3s, sums, counts);
    edge_mlp_kernel<<<EE / EPB, 256, 0, stream>>>(x, ei, ef, W1s, b1, W2s, b2, W3s, b3,
                                                  sums, counts);
    finalize_kernel<<<pblocks, 256, 0, stream>>>(x, sums, counts, (float*)d_out);
}
